// MultiHeadAttention_42726334660853
// MI455X (gfx1250) — matrix operand order repaired and checked by reading, not compiled, not run on hardware
//
#include <hip/hip_runtime.h>
#include <hip/hip_bf16.h>
#include <stdint.h>

#define BB 2
#define SS 2048
#define DM 512
#define HH 8
#define DEPTH 64

typedef __attribute__((ext_vector_type(16))) __bf16 v16bf;
typedef __attribute__((ext_vector_type(8)))  float  v8f;
typedef __attribute__((ext_vector_type(4)))  unsigned int u32x4;
typedef __attribute__((ext_vector_type(8)))  int i32x8;
typedef __attribute__((ext_vector_type(4)))  int i32x4;

#if __has_builtin(__builtin_amdgcn_tensor_load_to_lds) && \
    __has_builtin(__builtin_amdgcn_s_wait_tensorcnt)
#define HAVE_TDM 1
#else
#define HAVE_TDM 0
#endif

union Frag {
    v16bf v;
    uint4 q[2];
};

__device__ __forceinline__ v8f wmma_bf16(const Frag& a, const Frag& b, v8f c) {
    // (neg_a, A, neg_b, B, c_mod, C, reuse_a, reuse_b)
    return __builtin_amdgcn_wmma_f32_16x16x32_bf16(false, a.v, false, b.v,
                                                   (short)0, c, false, false);
}

// Async global->LDS 16-byte copy (gfx1250 GLOBAL_LOAD_ASYNC_TO_LDS_B128).
// LDS destination address = low 32 bits of the flat shared pointer.
__device__ __forceinline__ void async_copy_b128(void* lds_dst, const void* g_src) {
    unsigned ldsoff = (unsigned)(uintptr_t)lds_dst;
    asm volatile("global_load_async_to_lds_b128 %0, %1, off"
                 :: "v"(ldsoff), "v"(g_src) : "memory");
}
__device__ __forceinline__ void wait_async0() {
    asm volatile("s_wait_asynccnt 0" ::: "memory");
}

#if HAVE_TDM
// Tensor Data Mover: DMA a 2-D tile of bf16 (rows x rowlen) from global to LDS.
// D# packing per CDNA5 ISA ch.8 (group0 128b, group1 256b; groups 2/3 zero = 2-D).
// This toolchain's builtin is the 6-arg variant: (g0, g1, g2, g3, i32x8, cpol).
__device__ __forceinline__ void tdm_load_2d_bf16(void* lds_dst, const void* g_src,
                                                 unsigned rows, unsigned rowlen,
                                                 unsigned row_stride)
{
    const unsigned lds = (unsigned)(uintptr_t)lds_dst;
    const unsigned long long ga = (unsigned long long)(uintptr_t)g_src;
    u32x4 g0;
    g0[0] = 1u;                                        // count=1, user mode
    g0[1] = lds;                                       // lds_addr [63:32]
    g0[2] = (unsigned)(ga & 0xFFFFFFFFu);              // global_addr[31:0]
    g0[3] = (unsigned)((ga >> 32) & 0x1FFFFFFu)        // global_addr[56:32]
          | 0x80000000u;                               // type=2 ("image") [127:126]
    i32x8 g1;
    g1[0] = (int)(1u << 16);                           // data_size=1 (2B), wg_mask=0
    g1[1] = (int)((rowlen & 0xFFFFu) << 16);           // tensor_dim0[15:0] @bit48
    g1[2] = (int)(((rowlen >> 16) & 0xFFFFu)           // tensor_dim0[31:16]
          | ((rows & 0xFFFFu) << 16));                 // tensor_dim1[15:0] @bit80
    g1[3] = (int)(((rows >> 16) & 0xFFFFu)             // tensor_dim1[31:16]
          | ((rowlen & 0xFFFFu) << 16));               // tile_dim0 @bit112
    g1[4] = (int)(rows & 0xFFFFu);                     // tile_dim1 @bit128, tile_dim2=0
    g1[5] = (int)row_stride;                           // tensor_dim0_stride[31:0] @bit160
    g1[6] = 0;                                         // stride[47:32]=0, dim1_stride lo=0
    g1[7] = 0;                                         // dim1_stride hi=0
    const i32x4 z4 = {0, 0, 0, 0};
    const i32x8 z8 = {0, 0, 0, 0, 0, 0, 0, 0};
    __builtin_amdgcn_tensor_load_to_lds(g0, g1, z4, z4, z8, 0);
}
#endif

// ---------------------------------------------------------------------------
// GEMM: Y = X[4096x512] @ W^T[512x512] + bias
//   IN_F32:    X is fp32 (else bf16, staged via async-to-LDS)
//   HEADS_OUT: store bf16 split-head layout [B,H,S,64] (else fp32 [4096][512])
// ---------------------------------------------------------------------------
template<bool IN_F32, bool HEADS_OUT>
__global__ __launch_bounds__(256)
void gemm512(const void* __restrict__ Xv, const float* __restrict__ W,
             const float* __restrict__ bias, void* __restrict__ Yv)
{
    __shared__ __bf16 lA[128 * 32];
    __shared__ __bf16 lB[64 * 32];

    const int tid  = threadIdx.x;
    const int lane = tid & 31;
    const int w    = tid >> 5;
    const int half = lane >> 4;
    const int l16  = lane & 15;
    const int M0   = blockIdx.y * 128;
    const int N0   = blockIdx.x * 64;

    v8f acc[4];
    for (int n = 0; n < 4; ++n)
        for (int i = 0; i < 8; ++i) acc[n][i] = 0.f;

    for (int kt = 0; kt < 16; ++kt) {
        const int k0 = kt * 32;
        if (IN_F32) {
            const float* X = (const float*)Xv;
#pragma unroll
            for (int i = 0; i < 4; ++i) {
                int c = tid + i * 256;
                int row = c >> 3, kc = (c & 7) * 4;
                const float4 f = *(const float4*)&X[(size_t)(M0 + row) * DM + k0 + kc];
                __bf16* p = &lA[row * 32 + kc];
                p[0] = (__bf16)f.x; p[1] = (__bf16)f.y;
                p[2] = (__bf16)f.z; p[3] = (__bf16)f.w;
            }
        } else {
            const __bf16* X = (const __bf16*)Xv;
            int c = tid;
            int row = c >> 1, part = (c & 1) * 16;
            async_copy_b128(&lA[row * 32 + part],
                            &X[(size_t)(M0 + row) * DM + k0 + part]);
        }
#pragma unroll
        for (int i = 0; i < 2; ++i) {
            int c = tid + i * 256;
            int row = c >> 3, kc = (c & 7) * 4;
            const float4 f = *(const float4*)&W[(size_t)(N0 + row) * DM + k0 + kc];
            __bf16* p = &lB[row * 32 + kc];
            p[0] = (__bf16)f.x; p[1] = (__bf16)f.y;
            p[2] = (__bf16)f.z; p[3] = (__bf16)f.w;
        }
        if (!IN_F32) wait_async0();
        __syncthreads();

        Frag a;
        {
            int r  = w * 16 + l16;
            int c0 = half ? 8 : 0;
            a.q[0] = *(const uint4*)&lA[r * 32 + c0];
            a.q[1] = *(const uint4*)&lA[r * 32 + c0 + 16];
        }
#pragma unroll
        for (int nt = 0; nt < 4; ++nt) {
            Frag b;
            int n  = nt * 16 + l16;
            int kb = half * 8;
            b.q[0] = *(const uint4*)&lB[n * 32 + kb];
            b.q[1] = *(const uint4*)&lB[n * 32 + kb + 16];
            acc[nt] = wmma_bf16(a, b, acc[nt]);
        }
        __syncthreads();
    }

#pragma unroll
    for (int nt = 0; nt < 4; ++nt) {
        const int coln = N0 + nt * 16 + l16;
        const float bv = bias[coln];
#pragma unroll
        for (int v = 0; v < 8; ++v) {
            const int m = M0 + w * 16 + v + 8 * half;
            const float val = acc[nt][v] + bv;
            if (HEADS_OUT) {
                __bf16* Y = (__bf16*)Yv;
                int h  = coln >> 6, d = coln & 63;
                int b_ = m >> 11,  s = m & 2047;
                Y[(size_t)((b_ * HH + h) * SS + s) * DEPTH + d] = (__bf16)val;
            } else {
                float* Y = (float*)Yv;
                Y[(size_t)m * DM + coln] = val;
            }
        }
    }
}

// ---------------------------------------------------------------------------
// Fused flash-attention: per block = one (b,h) x 128 query rows.
// Q tile: TDM tensor_load_to_lds (fallback: async). K: async global->LDS.
// ---------------------------------------------------------------------------
__global__ __launch_bounds__(256)
void attn_kernel(const __bf16* __restrict__ Qh, const __bf16* __restrict__ Kh,
                 const __bf16* __restrict__ Vh, const float* __restrict__ maskp,
                 __bf16* __restrict__ attnOut)
{
    __shared__ __bf16 lQ[128 * 64];
    __shared__ __bf16 lK[32 * 64];
    __shared__ __bf16 lVt[64 * 32];      // V transposed: [depth][key]
    __shared__ __bf16 lP[8 * 16 * 32];   // per-wave P tile buffer
    __shared__ float  lMq[128];
    __shared__ float  lMk[32];

    const int tid  = threadIdx.x;
    const int lane = tid & 31;
    const int w    = tid >> 5;
    const int half = lane >> 4;
    const int l16  = lane & 15;

    const int QT = SS / 128;
    const int bh = blockIdx.x / QT;
    const int qt = blockIdx.x % QT;
    const int b  = bh / HH;
    const int h  = bh % HH;
    const int q0 = qt * 128;

    // ---- stage Q tile (128x64 bf16) + query mask ----
    {
        const __bf16* Qbase = Qh + (size_t)bh * SS * DEPTH + (size_t)q0 * DEPTH;
#if HAVE_TDM
        if (w == 0)
            tdm_load_2d_bf16(lQ, Qbase, /*rows=*/128, /*rowlen=*/DEPTH,
                             /*row_stride=*/DEPTH);
#else
#pragma unroll
        for (int i = 0; i < 4; ++i) {
            int c = tid + i * 256;
            int row = c >> 3, p = (c & 7) * 8;
            async_copy_b128(&lQ[row * 64 + p], &Qbase[row * DEPTH + p]);
        }
#endif
        if (tid < 128) lMq[tid] = maskp[b * SS + q0 + tid];
    }
#if HAVE_TDM
    __builtin_amdgcn_s_wait_tensorcnt(0);
#else
    wait_async0();
#endif
    __syncthreads();

    // Q A-fragments (two depth slices of 32) held in registers for all j
    Frag qf[2];
#pragma unroll
    for (int ds = 0; ds < 2; ++ds) {
        int r  = w * 16 + l16;
        int c0 = ds * 32 + (half ? 8 : 0);
        qf[ds].q[0] = *(const uint4*)&lQ[r * 64 + c0];
        qf[ds].q[1] = *(const uint4*)&lQ[r * 64 + c0 + 16];
    }
    float mq[8];
#pragma unroll
    for (int v = 0; v < 8; ++v) mq[v] = lMq[w * 16 + v + 8 * half];

    float rM[8], rL[8];
    v8f O[4];
#pragma unroll
    for (int v = 0; v < 8; ++v) { rM[v] = -1e30f; rL[v] = 0.f; }
    for (int n = 0; n < 4; ++n)
        for (int i = 0; i < 8; ++i) O[n][i] = 0.f;

    const __bf16* Kbase = Kh + (size_t)bh * SS * DEPTH;
    const __bf16* Vbase = Vh + (size_t)bh * SS * DEPTH;
    const float scale = 0.125f;

    for (int j = 0; j < SS / 32; ++j) {
        const int kb = j * 32;
        // ---- stage K block (async), V block transposed, key mask ----
        {
            int row = tid >> 3, p = (tid & 7) * 8;
            async_copy_b128(&lK[row * 64 + p],
                            &Kbase[(size_t)(kb + row) * DEPTH + p]);
            union { uint4 u; __bf16 e[8]; } vv;
            vv.u = *(const uint4*)&Vbase[(size_t)(kb + row) * DEPTH + p];
#pragma unroll
            for (int e = 0; e < 8; ++e) lVt[(p + e) * 32 + row] = vv.e[e];
            if (tid < 32) lMk[tid] = maskp[b * SS + kb + tid];
            if (j + 1 < SS / 32) {
                __builtin_prefetch(&Kbase[(size_t)(kb + 32 + row) * DEPTH + p], 0, 1);
                __builtin_prefetch(&Vbase[(size_t)(kb + 32 + row) * DEPTH + p], 0, 1);
            }
        }
        wait_async0();
        __syncthreads();

        // ---- logits: S = Q(16x64) @ K^T(64x32), two 16-key sub-tiles ----
        float lt[2][8];
#pragma unroll
        for (int nt = 0; nt < 2; ++nt) {
            v8f acc;
            for (int i = 0; i < 8; ++i) acc[i] = 0.f;
            int n   = nt * 16 + l16;
            int kdb = half * 8;
#pragma unroll
            for (int ds = 0; ds < 2; ++ds) {
                Frag bf;
                bf.q[0] = *(const uint4*)&lK[n * 64 + ds * 32 + kdb];
                bf.q[1] = *(const uint4*)&lK[n * 64 + ds * 32 + kdb + 16];
                acc = wmma_bf16(qf[ds], bf, acc);
            }
            float mk = lMk[n];
#pragma unroll
            for (int v = 0; v < 8; ++v) {
                float pen = fminf(mq[v] + mk, 1.f) * -1e9f;
                float x = acc[v] * scale + pen;
                lt[nt][v] = fminf(fmaxf(x, -50.f), 50.f);
            }
        }

        // ---- online softmax ----
        float alpha[8];
#pragma unroll
        for (int v = 0; v < 8; ++v) {
            float rv = fmaxf(lt[0][v], lt[1][v]);
#pragma unroll
            for (int sm = 1; sm <= 8; sm <<= 1)
                rv = fmaxf(rv, __shfl_xor(rv, sm, 32));
            float Mn = fmaxf(rM[v], rv);
            alpha[v] = __expf(rM[v] - Mn);
            rM[v] = Mn;
            lt[0][v] = __expf(lt[0][v] - Mn);
            lt[1][v] = __expf(lt[1][v] - Mn);
            float sv = lt[0][v] + lt[1][v];
#pragma unroll
            for (int sm = 1; sm <= 8; sm <<= 1)
                sv += __shfl_xor(sv, sm, 32);
            rL[v] = alpha[v] * rL[v] + sv;
        }
#pragma unroll
        for (int n = 0; n < 4; ++n)
#pragma unroll
            for (int v = 0; v < 8; ++v) O[n][v] *= alpha[v];

        // ---- P: C-layout -> LDS -> A-layout fragment ----
        __bf16* pb = &lP[w * 16 * 32];
#pragma unroll
        for (int nt = 0; nt < 2; ++nt)
#pragma unroll
            for (int v = 0; v < 8; ++v)
                pb[(v + 8 * half) * 32 + nt * 16 + l16] = (__bf16)lt[nt][v];
        asm volatile("s_wait_dscnt 0" ::: "memory");
        Frag pf;
        {
            int c0 = half ? 8 : 0;
            pf.q[0] = *(const uint4*)&pb[l16 * 32 + c0];
            pf.q[1] = *(const uint4*)&pb[l16 * 32 + c0 + 16];
        }
        // ---- O += P(16x32) @ V(32x64) ----
#pragma unroll
        for (int dt = 0; dt < 4; ++dt) {
            Frag bv;
            int d  = dt * 16 + l16;
            int kk = half * 8;
            bv.q[0] = *(const uint4*)&lVt[d * 32 + kk];
            bv.q[1] = *(const uint4*)&lVt[d * 32 + kk + 16];
            O[dt] = wmma_bf16(pf, bv, O[dt]);
        }
        __syncthreads();
    }

    // ---- normalize + store merged heads (bf16) ----
    float inv[8];
#pragma unroll
    for (int v = 0; v < 8; ++v) inv[v] = 1.f / rL[v];
#pragma unroll
    for (int dt = 0; dt < 4; ++dt) {
#pragma unroll
        for (int v = 0; v < 8; ++v) {
            int m    = b * SS + q0 + w * 16 + v + 8 * half;
            int coln = h * DEPTH + dt * 16 + l16;
            attnOut[(size_t)m * DM + coln] = (__bf16)(O[dt][v] * inv[v]);
        }
    }
}

// ---------------------------------------------------------------------------
extern "C" void kernel_launch(void* const* d_in, const int* in_sizes, int n_in,
                              void* d_out, int out_size, void* d_ws, size_t ws_size,
                              hipStream_t stream)
{
    (void)in_sizes; (void)n_in; (void)out_size; (void)ws_size;
    const float* v_in = (const float*)d_in[0];
    const float* k_in = (const float*)d_in[1];
    const float* q_in = (const float*)d_in[2];
    const float* mask = (const float*)d_in[3];
    const float* wq_w = (const float*)d_in[4];
    const float* wq_b = (const float*)d_in[5];
    const float* wk_w = (const float*)d_in[6];
    const float* wk_b = (const float*)d_in[7];
    const float* wv_w = (const float*)d_in[8];
    const float* wv_b = (const float*)d_in[9];
    const float* dw   = (const float*)d_in[10];
    const float* db   = (const float*)d_in[11];

    char* ws = (char*)d_ws;
    __bf16* qh = (__bf16*)(ws);
    __bf16* kh = (__bf16*)(ws + (4u << 20));
    __bf16* vh = (__bf16*)(ws + (8u << 20));
    __bf16* at = (__bf16*)(ws + (12u << 20));

    dim3 grid(DM / 64, (BB * SS) / 128), blk(256);
    gemm512<true, true><<<grid, blk, 0, stream>>>(q_in, wq_w, wq_b, qh);
    gemm512<true, true><<<grid, blk, 0, stream>>>(k_in, wk_w, wk_b, kh);
    gemm512<true, true><<<grid, blk, 0, stream>>>(v_in, wv_w, wv_b, vh);

    attn_kernel<<<dim3(BB * HH * (SS / 128)), blk, 0, stream>>>(qh, kh, vh, mask, at);

    gemm512<false, false><<<grid, blk, 0, stream>>>(at, dw, db, d_out);
}
